// AutoencoderModel_20255065768639
// MI455X (gfx1250) — matrix operand order repaired and checked by reading, not compiled, not run on hardware
//
#include <hip/hip_runtime.h>
#include <hip/hip_bf16.h>

typedef __attribute__((ext_vector_type(16))) _Float16 v16h;
typedef __attribute__((ext_vector_type(8)))  _Float16 v8h;
typedef __attribute__((ext_vector_type(8)))  float    v8f;
typedef __attribute__((ext_vector_type(4)))  int      v4i;
typedef __attribute__((address_space(1)))    v4i      gv4i;  // global int4
typedef __attribute__((address_space(3)))    v4i      sv4i;  // LDS int4

#define BN_EPS 1e-4f

// Async global->LDS staging (CDNA5 GLOBAL_LOAD_ASYNC_TO_LDS_B128, ASYNCcnt),
// guarded so the file still compiles if this toolchain lacks the builtin.
#if __has_builtin(__builtin_amdgcn_global_load_async_to_lds_b128)
#define USE_ASYNC_LDS 1
#else
#define USE_ASYNC_LDS 0
#endif

// ---------------------------------------------------------------------------
// WMMA K-loop: accumulate C += A(16x32 f16) * B(32x16 f16) over nck chunks.
// A per-lane: two contiguous 8-half (16B) chunks at K offsets kbase, kbase+16
//   (kbase = 0 for lanes 0-15, 8 for lanes 16-31) — matches ISA 16-bit A 16x32.
// B per-lane: two contiguous 8-half chunks at K offsets bko, bko+16 (bko = 0 / 8) — matches A.
// Works for both global and LDS `ap` (addrspace inferred -> ds_load_b128).
// ---------------------------------------------------------------------------
__device__ __forceinline__ void wmma_kloop(const _Float16* __restrict__ ap,
                                           const _Float16* __restrict__ bp,
                                           bool av, int nck, int kbase, int bko,
                                           v8f& c)
{
  for (int ck = 0; ck < nck; ++ck) {
    v8h a0 = {}, a1 = {};
    if (av) {
      a0 = *reinterpret_cast<const v8h*>(ap + ck * 32 + kbase);
      a1 = *reinterpret_cast<const v8h*>(ap + ck * 32 + kbase + 16);
    }
    v16h A = __builtin_shufflevector(a0, a1, 0,1,2,3,4,5,6,7,8,9,10,11,12,13,14,15);
    v8h b0 = *reinterpret_cast<const v8h*>(bp + ck * 32 + bko);
    v8h b1 = *reinterpret_cast<const v8h*>(bp + ck * 32 + bko + 16);
    v16h B = __builtin_shufflevector(b0, b1, 0,1,2,3,4,5,6,7,8,9,10,11,12,13,14,15);
    c = __builtin_amdgcn_wmma_f32_16x16x32_f16(false, A, false, B, (short)0, c,
                                               false, false);
  }
}

// ---------------------------------------------------------------------------
// LDS-staged 3x3x3 stride-1 SAME conv (the dominant FLOP path).
// Block stages the 3x3x18 x Cin halo patch once — via async-to-LDS when the
// toolchain exposes it (no VGPR round-trip, tracked by ASYNCcnt), else
// global_load_b128 + ds_store_b128 — then each wave computes one
// 16-voxel x 16-cout WMMA tile reading A from LDS (ds_load_b128).
// Requires Wo % 16 == 0 so every tile lies in one W-row.
// Up to 8 waves/block so each patch is staged exactly once per voxel tile.
// ---------------------------------------------------------------------------
__global__ void __launch_bounds__(256) k_subm3_wmma(
    const _Float16* __restrict__ in, const _Float16* __restrict__ w,
    const float* __restrict__ mask, float* __restrict__ out,
    int D, int H, int W, int Cin, int Cout, int acc)
{
  extern __shared__ _Float16 lds[];
  const long vox0 = (long)blockIdx.x * 16;
  const int  w0 = (int)(vox0 % W);
  const long t1 = vox0 / W;
  const int  oh = (int)(t1 % H);
  const int  od = (int)(t1 / H);

  // cooperative stage of the 3 x 3 x (16+2) x Cin patch, zero-filled halo
  const int c8n   = Cin >> 3;
  const int units = 162 * c8n;           // 3*3*18 voxels, 8 halfs per unit
  for (int u = threadIdx.x; u < units; u += blockDim.x) {
    const int c8 = u % c8n;
    const int pv = u / c8n;
    int q = pv;
    const int lw  = q % 18;  q /= 18;
    const int dy  = q % 3;
    const int dz  = q / 3;
    const int id = od + dz - 1, ih = oh + dy - 1, iw = w0 + lw - 1;
    const bool valid = (unsigned)id < (unsigned)D && (unsigned)ih < (unsigned)H &&
                       (unsigned)iw < (unsigned)W;
    const long soff = (((long)id * H + ih) * W + iw) * Cin + c8 * 8;
    _Float16* dptr = lds + (long)pv * Cin + c8 * 8;
#if USE_ASYNC_LDS
    if (valid) {
      __builtin_amdgcn_global_load_async_to_lds_b128(
          (gv4i*)(in + soff), (sv4i*)dptr, 0, 0);
    } else {
      v8h z = {};
      *reinterpret_cast<v8h*>(dptr) = z;
    }
#else
    v8h val = {};
    if (valid)
      val = *reinterpret_cast<const v8h*>(in + soff);
    *reinterpret_cast<v8h*>(dptr) = val;
#endif
  }
#if USE_ASYNC_LDS
#if __has_builtin(__builtin_amdgcn_s_wait_asynccnt)
  __builtin_amdgcn_s_wait_asynccnt(0);
#else
  asm volatile("s_wait_asynccnt 0" ::: "memory");
#endif
#endif
  __syncthreads();

  const int lane = threadIdx.x & 31;
  const int wave = threadIdx.x >> 5;
  const int co0  = (blockIdx.y * (blockDim.x >> 5) + wave) * 16;
  if (co0 < Cout) {                       // wave-uniform: EXEC stays all-1s
    const int m     = lane & 15;
    const int kbase = (lane < 16) ? 0 : 8;
    const int n     = lane & 15;
    const int bko   = (lane < 16) ? 0 : 8;
    const int nck   = Cin >> 5;
    v8f c = {};
    for (int t = 0; t < 27; ++t) {
      const int kd = t / 9, kh = (t / 3) % 3, kw = t % 3;
      const _Float16* ap = lds + (long)((kd * 3 + kh) * 18 + (m + kw)) * Cin;
      const _Float16* bp = w + ((long)t * Cout + co0 + n) * Cin;
      if (t < 26)                        // hide next tap's weight fetch
        __builtin_prefetch(w + ((long)(t + 1) * Cout + co0 + n) * Cin, 0, 0);
      wmma_kloop(ap, bp, true, nck, kbase, bko, c);
    }
    const int mh = (lane >> 4) << 3;
    for (int r = 0; r < 8; ++r) {
      long vv = vox0 + mh + r;           // W%16==0 -> always in range
      float val = c[r] * mask[vv];
      long o = vv * Cout + co0 + n;
      out[o] = acc ? (out[o] + val) : val;
    }
  }
}

// ---------------------------------------------------------------------------
// Generic KxKxK conv (implicit GEMM, direct global A loads). Covers the small
// deep levels (W < 16), 2x2x2 down conv (s=2,pad=0) and 1x1x1 NiN (K=1).
// Weights f16, layout [tap][co][cin_pad32]. Output f32, stride = Cout.
// ---------------------------------------------------------------------------
__global__ void __launch_bounds__(128) k_conv_wmma(
    const _Float16* __restrict__ in, const _Float16* __restrict__ w,
    const float* __restrict__ mask, float* __restrict__ out,
    int Di, int Hi, int Wi, int Do, int Ho, int Wo,
    int Cin, int Cout, int K, int stride, int pad, int acc)
{
  const int lane = threadIdx.x & 31;
  const int wave = threadIdx.x >> 5;
  const int co0  = (blockIdx.y * (blockDim.x >> 5) + wave) * 16;
  if (co0 >= Cout) return;                    // wave-uniform

  const long Nvox = (long)Do * Ho * Wo;
  const long vox0 = (long)blockIdx.x * 16;
  const int  m    = lane & 15;
  const long vm   = vox0 + m;
  const bool mrow = vm < Nvox;
  int od = 0, oh = 0, ow = 0;
  if (mrow) { long t = vm; ow = (int)(t % Wo); t /= Wo; oh = (int)(t % Ho); od = (int)(t / Ho); }

  const int kbase = (lane < 16) ? 0 : 8;
  const int n     = lane & 15;
  const int bko   = (lane < 16) ? 0 : 8;
  const int taps  = K * K * K;
  const int nck   = Cin >> 5;

  v8f c = {};
  for (int t = 0; t < taps; ++t) {
    const int kd = t / (K * K), kh = (t / K) % K, kw = t % K;
    const int id = od * stride + kd - pad;
    const int ih = oh * stride + kh - pad;
    const int iw = ow * stride + kw - pad;
    const bool av = mrow && (unsigned)id < (unsigned)Di &&
                    (unsigned)ih < (unsigned)Hi && (unsigned)iw < (unsigned)Wi;
    const _Float16* ap = in + (((long)id * Hi + ih) * Wi + iw) * Cin;
    const _Float16* bp = w + ((long)t * Cout + co0 + n) * Cin;
    wmma_kloop(ap, bp, av, nck, kbase, bko, c);
  }

  const int mh = (lane >> 4) << 3;
  for (int r = 0; r < 8; ++r) {
    long vv = vox0 + mh + r;
    if (vv < Nvox) {
      float val = c[r];
      if (mask) val *= mask[vv];
      long o = vv * Cout + co0 + n;
      out[o] = acc ? (out[o] + val) : val;
    }
  }
}

// ---------------------------------------------------------------------------
// Transpose conv (kernel 2, stride 2, VALID): each fine output voxel receives
// exactly one tap = its parity. Tile 16 same-parity outputs along W so the
// whole wave shares one tap -> a clean 16x16 WMMA GEMM over Cin.
// ---------------------------------------------------------------------------
__global__ void __launch_bounds__(128) k_upconv_wmma(
    const _Float16* __restrict__ in, const _Float16* __restrict__ w,
    const float* __restrict__ mask, float* __restrict__ out,
    int Dc, int Hc, int Wc, int Cin, int Cout, int Cstride, int coff)
{
  const int lane = threadIdx.x & 31;
  const int wave = threadIdx.x >> 5;
  const int co0  = (blockIdx.y * (blockDim.x >> 5) + wave) * 16;
  if (co0 >= Cout) return;

  const int Hf = 2 * Hc, Wf = 2 * Wc;
  const int nWt = (Wc + 15) >> 4;
  int bx = blockIdx.x;
  const int px  = bx & 1;  bx >>= 1;
  const int wt  = bx % nWt; bx /= nWt;
  const int ohf = bx % Hf;
  const int odf = bx / Hf;
  const int dc = odf >> 1, hc = ohf >> 1;
  const int tap = ((odf & 1) * 2 + (ohf & 1)) * 2 + px;

  const int m  = lane & 15;
  const int wc = wt * 16 + m;
  const bool av = wc < Wc;
  const int kbase = (lane < 16) ? 0 : 8;
  const int n     = lane & 15;
  const int bko   = (lane < 16) ? 0 : 8;
  const int nck   = Cin >> 5;

  const _Float16* ap = in + (((long)dc * Hc + hc) * Wc + wc) * Cin;
  const _Float16* bp = w + ((long)tap * Cout + co0 + n) * Cin;

  v8f c = {};
  wmma_kloop(ap, bp, av, nck, kbase, bko, c);

  const int mh = (lane >> 4) << 3;
  for (int r = 0; r < 8; ++r) {
    int wcr = wt * 16 + mh + r;
    if (wcr < Wc) {
      long vv = ((long)odf * Hf + ohf) * Wf + 2 * wcr + px;
      out[vv * Cstride + coff + co0 + n] = c[r] * mask[vv];
    }
  }
}

// ---------------------------------------------------------------------------
// Fused eval-mode BN + ReLU + mask, f32 -> f16 (with channel zero-padding).
// With gamma==nullptr it is a plain convert (optionally masked).
// ---------------------------------------------------------------------------
__global__ void k_bnrelu16(const float* __restrict__ x,
                           const float* __restrict__ beta,
                           const float* __restrict__ gamma,
                           const float* __restrict__ mean,
                           const float* __restrict__ var,
                           const float* __restrict__ mask,
                           _Float16* __restrict__ out,
                           long nvox, int C, int Cp)
{
  long i = (long)blockIdx.x * blockDim.x + threadIdx.x;
  if (i >= nvox * Cp) return;
  int  c = (int)(i % Cp);
  long v = i / Cp;
  float y = 0.f;
  if (c < C) {
    y = x[v * C + c];
    if (gamma) {
      y = (y - mean[c]) * rsqrtf(var[c] + BN_EPS) * gamma[c] + beta[c];
      y = fmaxf(y, 0.f);
    }
    if (mask) y *= mask[v];
  }
  out[i] = (_Float16)y;
}

// f32 [tap][ci][co] -> f16 [tap][co][cip] (cip = ci rounded up to 32, zero pad)
__global__ void k_wcvt(const float* __restrict__ w, _Float16* __restrict__ o,
                       int taps, int ci, int cip, int co)
{
  long i = (long)blockIdx.x * blockDim.x + threadIdx.x;
  long n = (long)taps * co * cip;
  if (i >= n) return;
  int  k  = (int)(i % cip);
  long r  = i / cip;
  int  c_o = (int)(r % co);
  int  t   = (int)(r / co);
  o[i] = (k < ci) ? (_Float16)w[((long)t * ci + k) * co + c_o] : (_Float16)0.f;
}

__global__ void k_mask(const int* __restrict__ act, float* __restrict__ m, long n)
{
  long i = (long)blockIdx.x * blockDim.x + threadIdx.x;
  if (i < n) m[i] = act[i] ? 1.f : 0.f;
}

// coarse mask = max over 2^3 children
__global__ void k_downmask(const float* __restrict__ mf, float* __restrict__ mc,
                           int Dc, int Hc, int Wc)
{
  long i = (long)blockIdx.x * blockDim.x + threadIdx.x;
  long n = (long)Dc * Hc * Wc;
  if (i >= n) return;
  int wc = (int)(i % Wc); long t = i / Wc;
  int hc = (int)(t % Hc); int dc = (int)(t / Hc);
  int Hf = 2 * Hc, Wf = 2 * Wc;
  float m = 0.f;
  for (int dz = 0; dz < 2; ++dz)
    for (int dy = 0; dy < 2; ++dy)
      for (int dx = 0; dx < 2; ++dx)
        m = fmaxf(m, mf[((long)(2 * dc + dz) * Hf + (2 * hc + dy)) * Wf + (2 * wc + dx)]);
  mc[i] = m;
}

// copy channels into first half of a concat buffer
__global__ void k_copychan(const float* __restrict__ src, float* __restrict__ dst,
                           long nvox, int Csrc, int Cdst, int off)
{
  long i = (long)blockIdx.x * blockDim.x + threadIdx.x;
  if (i >= nvox * Csrc) return;
  int  c = (int)(i % Csrc);
  long v = i / Csrc;
  dst[v * Cdst + off + c] = src[i];
}

// final Linear(32 -> 8) + bias, masked (Cout=8 too small for a WMMA tile)
__global__ void k_linear(const _Float16* __restrict__ h, const float* __restrict__ wl,
                         const float* __restrict__ bl, const float* __restrict__ mask,
                         float* __restrict__ out, long nvox, int Ci, int Co)
{
  long i = (long)blockIdx.x * blockDim.x + threadIdx.x;
  if (i >= nvox * Co) return;
  int  co = (int)(i % Co);
  long v  = i / Co;
  float s = bl[co];
  for (int ci = 0; ci < Ci; ++ci) s += (float)h[v * Ci + ci] * wl[ci * Co + co];
  out[i] = s * mask[v];
}

// ---------------------------------------------------------------------------
// Host-side orchestration
// ---------------------------------------------------------------------------
static inline unsigned cdiv(long a, long b) { return (unsigned)((a + b - 1) / b); }

static void launch_conv(hipStream_t s, const _Float16* in, const _Float16* w,
                        const float* mask, float* out,
                        int Di, int Hi, int Wi, int Do, int Ho, int Wo,
                        int Cin, int Cout, int K, int stride, int pad, int acc)
{
  long nvox = (long)Do * Ho * Wo;
  int coutTiles = (Cout + 15) / 16;
  if (K == 3 && stride == 1 && (Wo % 16) == 0 && Cin <= 192 && mask) {
    int wpb = coutTiles < 8 ? coutTiles : 8;   // stage each patch exactly once
    dim3 grid(cdiv(nvox, 16), (unsigned)((coutTiles + wpb - 1) / wpb), 1);
    dim3 block(32 * wpb, 1, 1);
    size_t ldsBytes = (size_t)162 * Cin * sizeof(_Float16);  // 3*3*18 x Cin
    k_subm3_wmma<<<grid, block, ldsBytes, s>>>(in, w, mask, out,
                                               Do, Ho, Wo, Cin, Cout, acc);
  } else {
    int wpb = coutTiles < 4 ? coutTiles : 4;
    dim3 grid(cdiv(nvox, 16), (unsigned)((coutTiles + wpb - 1) / wpb), 1);
    dim3 block(32 * wpb, 1, 1);
    k_conv_wmma<<<grid, block, 0, s>>>(in, w, mask, out, Di, Hi, Wi, Do, Ho, Wo,
                                       Cin, Cout, K, stride, pad, acc);
  }
}

static void launch_upconv(hipStream_t s, const _Float16* in, const _Float16* w,
                          const float* mask, float* out,
                          int Dc, int Hc, int Wc, int Cin, int Cout,
                          int Cstride, int coff)
{
  int nWt = (Wc + 15) / 16;
  long nb = (long)(2 * Dc) * (2 * Hc) * nWt * 2;
  int coutTiles = (Cout + 15) / 16;
  int wpb = coutTiles < 4 ? coutTiles : 4;
  dim3 grid((unsigned)nb, (unsigned)((coutTiles + wpb - 1) / wpb), 1);
  dim3 block(32 * wpb, 1, 1);
  k_upconv_wmma<<<grid, block, 0, s>>>(in, w, mask, out, Dc, Hc, Wc, Cin, Cout,
                                       Cstride, coff);
}

static void launch_bn(hipStream_t s, const float* x, const float* const* bn,
                      const float* mask, _Float16* out, long nvox, int C, int Cp)
{
  long n = nvox * Cp;
  k_bnrelu16<<<cdiv(n, 256), 256, 0, s>>>(x, bn ? bn[0] : nullptr,
                                          bn ? bn[1] : nullptr,
                                          bn ? bn[2] : nullptr,
                                          bn ? bn[3] : nullptr,
                                          mask, out, nvox, C, Cp);
}

// res block: out = shortcut(X) + conv2(bnrelu(conv1(bnrelu(X))))
static void run_res(hipStream_t s, const float* X, float* T, float* O,
                    _Float16* H1, _Float16* H2,
                    const float* const* bn1, const _Float16* c1,
                    const float* const* bn2, const _Float16* c2,
                    const _Float16* nin, const float* mask,
                    int S, long V, int a, int b)
{
  if (nin) {
    launch_bn(s, X, nullptr, nullptr, H2, V, a, a);            // plain f16 copy of X
    launch_conv(s, H2, nin, nullptr, O, S, S, S, S, S, S, a, b, 1, 1, 0, 0);
  } else {
    (void)hipMemcpyAsync(O, X, (size_t)V * a * sizeof(float),
                         hipMemcpyDeviceToDevice, s);
  }
  launch_bn(s, X, bn1, mask, H1, V, a, a);
  launch_conv(s, H1, c1, mask, T, S, S, S, S, S, S, a, b, 3, 1, 1, 0);
  launch_bn(s, T, bn2, mask, H2, V, b, b);
  launch_conv(s, H2, c2, mask, O, S, S, S, S, S, S, b, b, 3, 1, 1, 1);
}

extern "C" void kernel_launch(void* const* d_in, const int* in_sizes, int n_in,
                              void* d_out, int out_size, void* d_ws, size_t ws_size,
                              hipStream_t stream)
{
  (void)in_sizes; (void)n_in; (void)out_size; (void)ws_size;

  // --- parse inputs (pytree order: dict keys sorted alphabetically) --------
  int ii = 0;
  auto F = [&]() { return (const float*)d_in[ii++]; };
  const float* x = F();                 // x [1,64,64,64,8]
  const float* b_lin = F();             // params.b_lin
  const float* bn_out[4];               // params.bn_out: beta,gamma,mean,var
  for (int j = 0; j < 4; ++j) bn_out[j] = F();
  struct RawRes { const float* bn1[4]; const float* bn2[4];
                  const float* c1; const float* c2; const float* nin; };
  struct RawLv  { const float* down_bn[4]; const float* down_w;
                  RawRes post, pre; const float* up_bn[4]; const float* up_w; };
  RawLv rl[5] = {};
  for (int l = 0; l < 5; ++l) {
    if (l < 4) {  // keys: down_bn, down_w, post, pre, up_bn, up_w
      for (int j = 0; j < 4; ++j) rl[l].down_bn[j] = F();
      rl[l].down_w = F();
      for (int j = 0; j < 4; ++j) rl[l].post.bn1[j] = F();
      for (int j = 0; j < 4; ++j) rl[l].post.bn2[j] = F();
      rl[l].post.c1 = F(); rl[l].post.c2 = F(); rl[l].post.nin = F();
      for (int j = 0; j < 4; ++j) rl[l].pre.bn1[j] = F();
      for (int j = 0; j < 4; ++j) rl[l].pre.bn2[j] = F();
      rl[l].pre.c1 = F(); rl[l].pre.c2 = F(); rl[l].pre.nin = nullptr;
      for (int j = 0; j < 4; ++j) rl[l].up_bn[j] = F();
      rl[l].up_w = F();
    } else {      // deepest level: pre only
      for (int j = 0; j < 4; ++j) rl[l].pre.bn1[j] = F();
      for (int j = 0; j < 4; ++j) rl[l].pre.bn2[j] = F();
      rl[l].pre.c1 = F(); rl[l].pre.c2 = F(); rl[l].pre.nin = nullptr;
    }
  }
  const float* w_in  = F();
  const float* w_lin = F();
  const int*   active = (const int*)d_in[ii++];

  // --- workspace layout ----------------------------------------------------
  char* ws = (char*)d_ws; size_t off = 0;
  auto alloc = [&](size_t b) -> void* {
    void* p = ws + off; off = (off + b + 255) & ~(size_t)255; return p;
  };
  const int  planes[5] = {32, 64, 96, 128, 160};
  const long V0 = 64L * 64 * 64;
  float*     A_ = (float*)alloc((size_t)V0 * 64 * 4);   // concat / encoder temp
  float*     Bt = (float*)alloc((size_t)V0 * 32 * 4);   // conv1 temp
  float*     Ct = (float*)alloc((size_t)V0 * 32 * 4);   // decoder output
  _Float16*  H1 = (_Float16*)alloc((size_t)V0 * 64 * 2);
  _Float16*  H2 = (_Float16*)alloc((size_t)V0 * 64 * 2);
  float* mk[5]; float* ft[4];
  for (int l = 0; l < 5; ++l) { int S = 64 >> l; mk[l] = (float*)alloc((size_t)S * S * S * 4); }
  for (int l = 0; l < 4; ++l) { int S = 64 >> l; ft[l] = (float*)alloc((size_t)S * S * S * planes[l] * 4); }

  // --- convert all conv weights to f16 [tap][co][ci_pad32] -----------------
  auto cvtw = [&](const float* src, int taps, int ci, int co) -> const _Float16* {
    int cip = (ci + 31) & ~31;
    _Float16* dst = (_Float16*)alloc((size_t)taps * co * cip * 2);
    long n = (long)taps * co * cip;
    k_wcvt<<<cdiv(n, 256), 256, 0, stream>>>(src, dst, taps, ci, cip, co);
    return dst;
  };
  const _Float16* w_in16 = cvtw(w_in, 27, 8, 32);
  struct LvW { const _Float16 *down, *up, *pre1, *pre2, *post1, *post2, *nin; };
  LvW lw[5] = {};
  for (int l = 0; l < 5; ++l) {
    int p = planes[l];
    lw[l].pre1 = cvtw(rl[l].pre.c1, 27, p, p);
    lw[l].pre2 = cvtw(rl[l].pre.c2, 27, p, p);
    if (l < 4) {
      lw[l].down  = cvtw(rl[l].down_w, 8, p, planes[l + 1]);
      lw[l].up    = cvtw(rl[l].up_w,   8, planes[l + 1], p);
      lw[l].post1 = cvtw(rl[l].post.c1, 27, 2 * p, p);
      lw[l].post2 = cvtw(rl[l].post.c2, 27, p, p);
      lw[l].nin   = cvtw(rl[l].post.nin, 1, 2 * p, p);
    }
  }

  // --- masks per level -----------------------------------------------------
  k_mask<<<cdiv(V0, 256), 256, 0, stream>>>(active, mk[0], V0);
  for (int l = 1; l < 5; ++l) {
    int Sc = 64 >> l; long n = (long)Sc * Sc * Sc;
    k_downmask<<<cdiv(n, 256), 256, 0, stream>>>(mk[l - 1], mk[l], Sc, Sc, Sc);
  }

  // --- input layer: (x*mask) -> f16 (pad 8->32 ch), then 3^3 conv 8->32 ----
  launch_bn(stream, x, nullptr, mk[0], H1, V0, 8, 32);
  launch_conv(stream, H1, w_in16, mk[0], A_, 64, 64, 64, 64, 64, 64, 32, 32, 3, 1, 1, 0);

  // --- encoder -------------------------------------------------------------
  for (int l = 0; l < 4; ++l) {
    int S = 64 >> l; long V = (long)S * S * S; int p = planes[l];
    run_res(stream, A_, Bt, ft[l], H1, H2,
            rl[l].pre.bn1, lw[l].pre1, rl[l].pre.bn2, lw[l].pre2,
            nullptr, mk[l], S, V, p, p);
    launch_bn(stream, ft[l], rl[l].down_bn, mk[l], H1, V, p, p);
    launch_conv(stream, H1, lw[l].down, mk[l + 1], A_,
                S, S, S, S / 2, S / 2, S / 2, p, planes[l + 1], 2, 2, 0, 0);
  }

  // --- bottleneck ----------------------------------------------------------
  run_res(stream, A_, Ct, Bt, H1, H2,
          rl[4].pre.bn1, lw[4].pre1, rl[4].pre.bn2, lw[4].pre2,
          nullptr, mk[4], 4, 64, 160, 160);
  float* cur = Bt;

  // --- decoder -------------------------------------------------------------
  for (int l = 3; l >= 0; --l) {
    int S = 64 >> l, Sc = S / 2;
    long V = (long)S * S * S, Vc = (long)Sc * Sc * Sc;
    int p = planes[l], pc = planes[l + 1];
    launch_bn(stream, cur, rl[l].up_bn, mk[l + 1], H1, Vc, pc, pc);
    { long n = V * (long)p;
      k_copychan<<<cdiv(n, 256), 256, 0, stream>>>(ft[l], A_, V, p, 2 * p, 0); }
    launch_upconv(stream, H1, lw[l].up, mk[l], A_, Sc, Sc, Sc, pc, p, 2 * p, p);
    run_res(stream, A_, Bt, Ct, H1, H2,
            rl[l].post.bn1, lw[l].post1, rl[l].post.bn2, lw[l].post2,
            lw[l].nin, mk[l], S, V, 2 * p, p);
    cur = Ct;
  }

  // --- output: BN-ReLU(32) then Linear 32->8 + bias, masked ----------------
  launch_bn(stream, cur, bn_out, mk[0], H1, V0, 32, 32);
  { long n = V0 * 8;
    k_linear<<<cdiv(n, 256), 256, 0, stream>>>(H1, w_lin, b_lin, mk[0],
                                               (float*)d_out, V0, 32, 8); }
}
